// LSTMAutoEncoder_12421045420047
// MI455X (gfx1250) — compile-verified, matrix operand order checked by reading, not run on hardware
//
#include <hip/hip_runtime.h>

typedef __attribute__((ext_vector_type(16))) _Float16 v16h;
typedef __attribute__((ext_vector_type(8)))  float    v8f;
typedef __attribute__((ext_vector_type(2)))  float    f2;

#define T_STEPS 256
#define HID 32

// gfx1250 has V_TANH_F32 (transcendental unit). Prefer the builtin; fall back to asm.
__device__ __forceinline__ float fast_tanh(float x) {
#if __has_builtin(__builtin_amdgcn_tanhf)
  return __builtin_amdgcn_tanhf(x);
#else
  float y;
  asm("v_tanh_f32 %0, %1" : "=v"(y) : "v"(x));
  return y;
#endif
}
// sigmoid of a gate whose pre-activation was already scaled by 0.5 (folded into weights):
// sigmoid(2*xh) = 0.5*tanh(xh) + 0.5  -> tanh + fma only
__device__ __forceinline__ float sig_prescaled(float xh) {
  return __builtin_fmaf(fast_tanh(xh), 0.5f, 0.5f);
}

// K column owned by packed dword v for lane-half hb (identical for A and B 16-bit layouts)
__device__ __forceinline__ int kcol(int v, int hb) {
  return (v < 4) ? (8 * hb + 2 * v) : (16 + 8 * hb + 2 * (v - 4));
}

// Compute one 16x16 gate tile: C init = bias + Wih*x (per-lane), then WMMA accumulate W_hh*h^T.
#define COMP_TILE(TL, DST)                                                            \
  {                                                                                   \
    v8f cc;                                                                           \
    _Pragma("unroll")                                                                 \
    for (int r = 0; r < 8; ++r)                                                       \
      cc[r] = __builtin_fmaf(bw[(TL) * 8 + r][1], xb, bw[(TL) * 8 + r][0]);           \
    cc = __builtin_amdgcn_wmma_f32_16x16x32_f16(false, A[(TL)], false, Bst,           \
                                                (short)0, cc, false, false);          \
    _Pragma("unroll")                                                                 \
    for (int r = 0; r < 8; ++r) (DST)[r] = cc[r];                                     \
  }

// ---------------- Encoder: LSTM(1 -> 32), keep final h ----------------
// One wave per 16-batch tile. A = W_hh (8 M-tiles of 16 gate rows x K=32) preloaded f16,
// pre-scaled by 0.5 for the sigmoid gates (i,f,o) so sigmoid costs tanh+fma only.
// B = h^T (K=32 x N=16 batch) f16, rebuilt per step purely per-lane (layouts align).
__global__ __launch_bounds__(32) void enc_kernel(
    const float* __restrict__ x, const float* __restrict__ Wih,
    const float* __restrict__ Whh, const float* __restrict__ bih,
    const float* __restrict__ bhh, float* __restrict__ hout /* [B][32] */) {
  const int lane = threadIdx.x;
  const int ln = lane & 15;   // N (batch-in-tile) for B/C/D; M (row-in-tile) for A
  const int hb = lane >> 4;   // lane half
  const long b0 = (long)blockIdx.x * 16;

  // Preload A = W_hh in f16 A-layout (constant over time). Tiles 0..3 (i,f) and 6..7 (o)
  // are sigmoid gates -> pre-scale by 0.5; tiles 4..5 (g, tanh) unscaled.
  v16h A[8];
#pragma unroll
  for (int m = 0; m < 8; ++m) {
    const float sc = (m == 4 || m == 5) ? 1.0f : 0.5f;
    const int row = 16 * m + ln;
#pragma unroll
    for (int v = 0; v < 8; ++v) {
      const int k0 = kcol(v, hb);
      A[m][2 * v]     = (_Float16)(sc * Whh[row * HID + k0]);
      A[m][2 * v + 1] = (_Float16)(sc * Whh[row * HID + k0 + 1]);
    }
  }

  // Per-lane (bias, Wih) f32 pairs for the 64 gate rows this lane owns in D layout,
  // with the same 0.5 pre-scale for sigmoid-gate rows.
  f2 bw[64];
#pragma unroll
  for (int tl = 0; tl < 8; ++tl) {
    const float sc = (tl == 4 || tl == 5) ? 1.0f : 0.5f;
#pragma unroll
    for (int r = 0; r < 8; ++r) {
      const int j = 16 * tl + r + 8 * hb;
      f2 p;
      p[0] = sc * (bih[j] + bhh[j]);
      p[1] = sc * Wih[j];
      bw[tl * 8 + r] = p;
    }
  }

  const float* xrow = x + (b0 + ln) * T_STEPS;

  v16h Bst;  // h state in f16 B-layout
#pragma unroll
  for (int v = 0; v < 16; ++v) Bst[v] = (_Float16)0.0f;
  float c0[8], c1[8];
#pragma unroll
  for (int r = 0; r < 8; ++r) { c0[r] = 0.0f; c1[r] = 0.0f; }

  float xb = xrow[0];
  for (int t = 0; t < T_STEPS; ++t) {
    const float xb_next = (t + 1 < T_STEPS) ? xrow[t + 1] : 0.0f;

    float ai0[8], ai1[8];
    {
      float g0[8], g1[8];
      COMP_TILE(0, g0); COMP_TILE(1, g1);
#pragma unroll
      for (int r = 0; r < 8; ++r) { ai0[r] = sig_prescaled(g0[r]); ai1[r] = sig_prescaled(g1[r]); }
    }
    {
      float g0[8], g1[8];
      COMP_TILE(2, g0); COMP_TILE(3, g1);
#pragma unroll
      for (int r = 0; r < 8; ++r) { c0[r] *= sig_prescaled(g0[r]); c1[r] *= sig_prescaled(g1[r]); }
    }
    {
      float g0[8], g1[8];
      COMP_TILE(4, g0); COMP_TILE(5, g1);
#pragma unroll
      for (int r = 0; r < 8; ++r) {
        c0[r] = __builtin_fmaf(ai0[r], fast_tanh(g0[r]), c0[r]);
        c1[r] = __builtin_fmaf(ai1[r], fast_tanh(g1[r]), c1[r]);
      }
    }
    float hn0[8], hn1[8];
    {
      float g0[8], g1[8];
      COMP_TILE(6, g0); COMP_TILE(7, g1);
#pragma unroll
      for (int r = 0; r < 8; ++r) {
        hn0[r] = sig_prescaled(g0[r]) * fast_tanh(c0[r]);
        hn1[r] = sig_prescaled(g1[r]) * fast_tanh(c1[r]);
      }
    }
    // Repack new h into B layout: pure per-lane (D half-split == B half-split)
#pragma unroll
    for (int v = 0; v < 4; ++v) {
      Bst[2 * v]           = (_Float16)hn0[2 * v];
      Bst[2 * v + 1]       = (_Float16)hn0[2 * v + 1];
      Bst[2 * (v + 4)]     = (_Float16)hn1[2 * v];
      Bst[2 * (v + 4) + 1] = (_Float16)hn1[2 * v + 1];
    }
    xb = xb_next;
  }

  // Store final h [16 batches][32] fp32
  float* hp = hout + (b0 + ln) * HID;
#pragma unroll
  for (int v = 0; v < 8; ++v) {
    const int k0 = kcol(v, hb);
    f2 o;
    o[0] = (float)Bst[2 * v];
    o[1] = (float)Bst[2 * v + 1];
    *(f2*)(hp + k0) = o;
  }
}

// ---------------- Decoder: LSTM(32 -> 1), input constant over time ----------------
// One lane per batch element. z = b + W_ih @ h hoisted out of the time loop.
// i/f/o pre-activations pre-scaled by 0.5 so sigmoid = tanh + fma.
__global__ __launch_bounds__(256) void dec_kernel(
    const float* __restrict__ hin, const float* __restrict__ Wih,
    const float* __restrict__ Whh, const float* __restrict__ bih,
    const float* __restrict__ bhh, float* __restrict__ out, int Btot) {
  const int b = blockIdx.x * blockDim.x + threadIdx.x;
  if (b >= Btot) return;

  float h[HID];
  const float* hp = hin + (long)b * HID;
#pragma unroll
  for (int k = 0; k < HID; ++k) h[k] = hp[k];

  float z[4], w[4];
#pragma unroll
  for (int i = 0; i < 4; ++i) {
    const float sc = (i == 2) ? 1.0f : 0.5f;  // gate order i,f,g,o; g uses plain tanh
    float acc = bih[i] + bhh[i];
#pragma unroll
    for (int k = 0; k < HID; ++k) acc = __builtin_fmaf(Wih[i * HID + k], h[k], acc);
    z[i] = sc * acc;
    w[i] = sc * Whh[i];  // dec_W_hh is [4][1]
  }

  float hd = 0.0f, cd = 0.0f;
  float* op = out + (long)b * T_STEPS;
  for (int t = 0; t < T_STEPS; ++t) {
    const float ig = sig_prescaled(__builtin_fmaf(w[0], hd, z[0]));
    const float fg = sig_prescaled(__builtin_fmaf(w[1], hd, z[1]));
    const float gg = fast_tanh(__builtin_fmaf(w[2], hd, z[2]));
    const float og = sig_prescaled(__builtin_fmaf(w[3], hd, z[3]));
    cd = __builtin_fmaf(fg, cd, ig * gg);
    hd = og * fast_tanh(cd);
    op[t] = hd;
  }
}

extern "C" void kernel_launch(void* const* d_in, const int* in_sizes, int n_in,
                              void* d_out, int out_size, void* d_ws, size_t ws_size,
                              hipStream_t stream) {
  const float* x    = (const float*)d_in[0];
  const float* eWih = (const float*)d_in[1];
  const float* eWhh = (const float*)d_in[2];
  const float* ebih = (const float*)d_in[3];
  const float* ebhh = (const float*)d_in[4];
  const float* dWih = (const float*)d_in[5];
  const float* dWhh = (const float*)d_in[6];
  const float* dbih = (const float*)d_in[7];
  const float* dbhh = (const float*)d_in[8];

  const int Btot = in_sizes[0] / T_STEPS;  // x is [B, T, 1]
  float* hbuf = (float*)d_ws;              // [B][32] fp32 final encoder hidden

  enc_kernel<<<Btot / 16, 32, 0, stream>>>(x, eWih, eWhh, ebih, ebhh, hbuf);
  dec_kernel<<<(Btot + 255) / 256, 256, 0, stream>>>(hbuf, dWih, dWhh, dbih, dbhh,
                                                     (float*)d_out, Btot);
}
